// PNNLayer_75368086110679
// MI455X (gfx1250) — compile-verified, matrix operand order checked by reading, not run on hardware
//
#include <hip/hip_runtime.h>
#include <math.h>

// ---------------------------------------------------------------------------
// PNN layer: per (m,k) slice, z = x @ (W*pos)^T + invx @ (W*neg)^T, then tanh
// activation.  E=8192, J=514 (padded to 544), O=512, MK=8.
// F16 WMMA (v_wmma_f32_16x16x32_f16) with fp32 accumulate.
// ---------------------------------------------------------------------------

#define MKC  8
#define EC   8192
#define NIN  512
#define JC   514
#define JP   544          // padded K (17 * 32)
#define OC   512
#define KSTEPS (JP / 32)  // 17

typedef _Float16 v16h __attribute__((ext_vector_type(16)));
typedef _Float16 h8   __attribute__((ext_vector_type(8)));
typedef float    v8f  __attribute__((ext_vector_type(8)));
typedef float    f8   __attribute__((ext_vector_type(8)));

__device__ __forceinline__ v16h load_frag16(const _Float16* __restrict__ p) {
    // Per-lane fragment: two contiguous 16-byte chunks (K chunk and K chunk+16)
    h8 lo = *(const h8*)(p);
    h8 hi = *(const h8*)(p + 16);
    v16h f;
#pragma unroll
    for (int i = 0; i < 8; ++i) { f[i] = lo[i]; f[i + 8] = hi[i]; }
    return f;
}

// ---------------------------------------------------------------------------
// Phase 0: materialize invx = 0.104 - 0.899*tanh((x+0.056)*3.858) in f16,
// with x = [a | 1 | 0], last column zeroed, padded with zeros to JP.
// One wave per (mk,e) row.
// ---------------------------------------------------------------------------
__global__ __launch_bounds__(256)
void pnn_invx_kernel(const float* __restrict__ a, _Float16* __restrict__ invx) {
    const int lane = threadIdx.x & 31;
    const int wave = threadIdx.x >> 5;
    const size_t row = (size_t)blockIdx.x * 8 + wave;   // [0, MK*EC)
    const float* __restrict__ ar = a + row * NIN;
    _Float16* __restrict__ ir = invx + row * JP;
#pragma unroll 4
    for (int j = lane; j < JP; j += 32) {
        float v = 0.0f;
        if (j < JC - 1) {                        // j==513 is zeroed, pads zero
            float x = (j < NIN) ? ar[j] : 1.0f;  // j==512 is the ones column
            v = 0.104f - 0.899f * tanhf((x + 0.056f) * 3.858f);
        }
        ir[j] = (_Float16)v;
    }
}

// ---------------------------------------------------------------------------
// Phase 1: per (mk,o) row, straight-through theta, g = |theta_st * aging|,
// W = g / sum_j g, split by sign(theta_st) into Bpos/Bneg (f16, [row][JP]).
// One wave per row; wave32 butterfly reduction.
// ---------------------------------------------------------------------------
__global__ __launch_bounds__(256)
void pnn_weights_kernel(const float* __restrict__ theta,
                        const float* __restrict__ aging,
                        _Float16* __restrict__ bpos,
                        _Float16* __restrict__ bneg) {
    const int lane = threadIdx.x & 31;
    const int wave = threadIdx.x >> 5;
    const int row = blockIdx.x * 8 + wave;      // [0, MK*OC)
    const int o = row & (OC - 1);
    const float* __restrict__ th = theta + (size_t)o * JC;
    const float* __restrict__ ag = aging + (size_t)row * JC;

    float s = 0.0f;
    for (int j = lane; j < JC; j += 32) {
        float t  = th[j];
        float ts = (fabsf(t) < 0.01f) ? 0.0f : t;
        s += fabsf(ts * ag[j]);
    }
#pragma unroll
    for (int off = 16; off > 0; off >>= 1) s += __shfl_xor(s, off, 32);
    const float rcp = 1.0f / s;

    _Float16* __restrict__ bp = bpos + (size_t)row * JP;
    _Float16* __restrict__ bn = bneg + (size_t)row * JP;
    for (int j = lane; j < JP; j += 32) {
        float wp = 0.0f, wn = 0.0f;
        if (j < JC) {
            float t  = th[j];
            float ts = (fabsf(t) < 0.01f) ? 0.0f : t;
            float w  = fabsf(ts * ag[j]) * rcp;
            if (ts >= 0.0f) wp = w; else wn = w;
        }
        bp[j] = (_Float16)wp;
        bn[j] = (_Float16)wn;
    }
}

// ---------------------------------------------------------------------------
// Phase 2: WMMA GEMM.  Block = 256 threads (8 waves) -> 128(E) x 128(O) tile.
// Wave tile = 64(E) x 32(O) = 4x2 accumulators of 16x16.
// K loop over JP in 32-steps; step 16 (j>=512) is the constant columns.
// ---------------------------------------------------------------------------
__global__ __launch_bounds__(256)
void pnn_gemm_kernel(const float* __restrict__ a,
                     const _Float16* __restrict__ invx,
                     const _Float16* __restrict__ bpos,
                     const _Float16* __restrict__ bneg,
                     float* __restrict__ out) {
    const int lane   = threadIdx.x & 31;
    const int wave   = threadIdx.x >> 5;
    const int we     = wave & 1;                // E group within block
    const int wo     = wave >> 1;               // O group within block
    const int mk     = blockIdx.z;
    const int eTile  = blockIdx.x * 128 + we * 64;
    const int oTile  = blockIdx.y * 128 + wo * 32;
    const int hi16   = lane >> 4;               // 0: lanes 0-15, 1: lanes 16-31
    const int sel    = hi16 * 8;                // K chunk select within step
    const int lr     = lane & 15;               // row-in-fragment / col-in-frag

    v8f acc[4][2];
#pragma unroll
    for (int e = 0; e < 4; ++e)
#pragma unroll
        for (int o = 0; o < 2; ++o)
#pragma unroll
            for (int r = 0; r < 8; ++r) acc[e][o][r] = 0.0f;

    const size_t aBase   = (size_t)mk * EC * NIN;
    const size_t invBase = (size_t)mk * EC * JP;
    const size_t bBase   = (size_t)mk * OC * JP;

    for (int kk = 0; kk < KSTEPS; ++kk) {
        const int kBase = kk * 32;

        // ---- B fragments (Bpos / Bneg), stored [o][j] row-major ----
        v16h fbp[2], fbn[2];
#pragma unroll
        for (int oo = 0; oo < 2; ++oo) {
            const size_t boff = bBase + (size_t)(oTile + oo * 16 + lr) * JP + kBase + sel;
            fbp[oo] = load_frag16(bpos + boff);
            fbn[oo] = load_frag16(bneg + boff);
            if (kk + 1 < KSTEPS) {
                __builtin_prefetch(bpos + boff + 32, 0, 0);
                __builtin_prefetch(bneg + boff + 32, 0, 0);
            }
        }

        // ---- A fragments: invx (f16 from ws) and x (cast from fp32 a) ----
        v16h fax[4], fai[4];
#pragma unroll
        for (int ee = 0; ee < 4; ++ee) {
            const int eRow = eTile + ee * 16 + lr;
            const size_t ioff = invBase + (size_t)eRow * JP + kBase + sel;
            fai[ee] = load_frag16(invx + ioff);
            if (kk + 1 < KSTEPS) __builtin_prefetch(invx + ioff + 32, 0, 0);
            if (kk < 16) {
                // all j < 512 in this step: straight loads from a
                const float* ap = a + aBase + (size_t)eRow * NIN + kBase + sel;
                f8 lo = *(const f8*)(ap);
                f8 hi = *(const f8*)(ap + 16);
#pragma unroll
                for (int i = 0; i < 8; ++i) {
                    fax[ee][i]     = (_Float16)lo[i];
                    fax[ee][i + 8] = (_Float16)hi[i];
                }
            } else {
                // kBase == 512: constant columns (j=512 -> 1, j>=513 -> 0)
#pragma unroll
                for (int i = 0; i < 16; ++i) fax[ee][i] = (_Float16)0.0f;
                if (hi16 == 0) fax[ee][0] = (_Float16)1.0f;  // j == 512
            }
        }

        // ---- 16 WMMAs: acc += x*Bpos + invx*Bneg ----
#pragma unroll
        for (int ee = 0; ee < 4; ++ee) {
#pragma unroll
            for (int oo = 0; oo < 2; ++oo) {
                acc[ee][oo] = __builtin_amdgcn_wmma_f32_16x16x32_f16(
                    false, fax[ee], false, fbp[oo], (short)0, acc[ee][oo],
                    false, false);
                acc[ee][oo] = __builtin_amdgcn_wmma_f32_16x16x32_f16(
                    false, fai[ee], false, fbn[oo], (short)0, acc[ee][oo],
                    false, false);
            }
        }
    }

    // ---- Epilogue: activation + store ----
    const size_t outBase = (size_t)mk * EC * OC;
#pragma unroll
    for (int ee = 0; ee < 4; ++ee) {
#pragma unroll
        for (int oo = 0; oo < 2; ++oo) {
            const int oCol = oTile + oo * 16 + lr;
#pragma unroll
            for (int r = 0; r < 8; ++r) {
                const int eRow = eTile + ee * 16 + r + hi16 * 8;
                float z = acc[ee][oo][r];
                float y = 0.134f + 0.962f * tanhf((z - 0.183f) * 24.1f);
                out[outBase + (size_t)eRow * OC + oCol] = y;
            }
        }
    }
}

// ---------------------------------------------------------------------------
extern "C" void kernel_launch(void* const* d_in, const int* in_sizes, int n_in,
                              void* d_out, int out_size, void* d_ws, size_t ws_size,
                              hipStream_t stream) {
    const float* a     = (const float*)d_in[0];   // [2,4,8192,512]
    const float* theta = (const float*)d_in[1];   // [512,514]
    const float* aging = (const float*)d_in[2];   // [2,4,512,514]
    float* out = (float*)d_out;                   // [2,4,8192,512]

    char* ws = (char*)d_ws;
    const size_t invBytes = (size_t)MKC * EC * JP * sizeof(_Float16); // ~71.3 MB
    const size_t bCount   = (size_t)MKC * OC * JP;
    _Float16* invx = (_Float16*)ws;
    _Float16* bpos = (_Float16*)(ws + invBytes);
    _Float16* bneg = bpos + bCount;

    // Phase 0: invx in f16 (one wave per (mk,e) row)
    pnn_invx_kernel<<<(MKC * EC) / 8, 256, 0, stream>>>(a, invx);

    // Phase 1: normalized sign-split weights in f16 (one wave per (mk,o) row)
    pnn_weights_kernel<<<(MKC * OC) / 8, 256, 0, stream>>>(theta, aging, bpos, bneg);

    // Phase 2: WMMA GEMM + activation
    dim3 grid(EC / 128, OC / 128, MKC);
    pnn_gemm_kernel<<<grid, 256, 0, stream>>>(a, invx, bpos, bneg, out);
}
